// MoELayer_13649406066703
// MI455X (gfx1250) — matrix operand order repaired and checked by reading, not compiled, not run on hardware
//
#include <hip/hip_runtime.h>
#include <math.h>
#include <stdint.h>

// Problem constants (from reference)
#define BB 2
#define SS 2048
#define TT (BB * SS)     // 4096 tokens
#define DD 1024
#define EE 8
#define FF 4096
#define TOPK 2

// Tiling
#define TM 32            // tokens per block in FFN kernel
#define FC 128           // F-chunk per iteration
#define MAXBLK (TT / TM) // 128 token-blocks per expert (worst case)
#define XS_LD (DD + 8)   // padded LDS stride (bf16 elems) to avoid bank conflicts
#define HS_LD (FC + 8)

typedef __bf16 bf16;
typedef __attribute__((ext_vector_type(16))) bf16 v16bf;
typedef __attribute__((ext_vector_type(8)))  bf16 v8bf;
typedef __attribute__((ext_vector_type(4)))  bf16 v4bf;
typedef __attribute__((ext_vector_type(8)))  float v8f;

// sched_group_barrier masks
#define SG_WMMA   0x008   // MFMA/WMMA group
#define SG_VMREAD 0x020   // VMEM read group
#define SG_DSREAD 0x100   // DS read group

static __device__ __forceinline__ v16bf cat8(v8bf lo, v8bf hi) {
    return __builtin_shufflevector(lo, hi, 0,1,2,3,4,5,6,7,8,9,10,11,12,13,14,15);
}

static __device__ __forceinline__ v16bf wmma_bf16(v16bf a, v16bf b, v8f& c) {
    c = __builtin_amdgcn_wmma_f32_16x16x32_bf16(false, a, false, b, (short)0, c,
                                                false, false);
    return a;
}

// ---------------------------------------------------------------------------
// Tiled transpose + fp32->bf16 convert: src [R][C] fp32 -> dst [C][R] bf16.
// grid: (C/32, R/32, E), block: 256
// ---------------------------------------------------------------------------
__global__ void transpose_bf16_kernel(const float* __restrict__ src,
                                      bf16* __restrict__ dst, int R, int C) {
    __shared__ float tile[32][33];
    const int e = blockIdx.z;
    src += (size_t)e * R * C;
    dst += (size_t)e * R * C;
    const int c0 = blockIdx.x * 32;
    const int r0 = blockIdx.y * 32;
    const int tx = threadIdx.x & 31;
    const int ty = threadIdx.x >> 5;   // 0..7
    #pragma unroll
    for (int rr = ty; rr < 32; rr += 8)
        tile[rr][tx] = src[(size_t)(r0 + rr) * C + (c0 + tx)];
    __syncthreads();
    #pragma unroll
    for (int rr = ty; rr < 32; rr += 8)
        dst[(size_t)(c0 + rr) * R + (r0 + tx)] = (bf16)tile[tx][rr];
}

// ---------------------------------------------------------------------------
// Router: one wave32 per token. scores = x @ gate_w, top-2, softmax over top-2,
// atomic scatter into per-expert token lists.
// ---------------------------------------------------------------------------
__global__ void router_kernel(const float* __restrict__ x,
                              const float* __restrict__ gw,
                              int* __restrict__ cnt,
                              int* __restrict__ tok_idx,
                              float* __restrict__ tok_prob) {
    const int wave = (int)((blockIdx.x * blockDim.x + threadIdx.x) >> 5);
    const int lane = threadIdx.x & 31;
    if (wave >= TT) return;
    const float* xr = x + (size_t)wave * DD;

    float acc[EE];
    #pragma unroll
    for (int e = 0; e < EE; ++e) acc[e] = 0.f;

    for (int d = lane; d < DD; d += 32) {
        const float xv = xr[d];
        const float* g = gw + (size_t)d * EE;
        #pragma unroll
        for (int e = 0; e < EE; ++e) acc[e] += xv * g[e];
    }
    #pragma unroll
    for (int e = 0; e < EE; ++e) {
        float v = acc[e];
        #pragma unroll
        for (int off = 16; off > 0; off >>= 1) v += __shfl_down(v, off, 32);
        acc[e] = v;
    }
    if (lane == 0) {
        int i0 = 0; float v0 = acc[0];
        #pragma unroll
        for (int e = 1; e < EE; ++e) if (acc[e] > v0) { v0 = acc[e]; i0 = e; }
        int i1 = -1; float v1 = -INFINITY;
        #pragma unroll
        for (int e = 0; e < EE; ++e)
            if (e != i0 && acc[e] > v1) { v1 = acc[e]; i1 = e; }
        const float e1 = __expf(v1 - v0);
        const float inv = 1.f / (1.f + e1);
        const float p0 = inv, p1 = e1 * inv;
        int p = atomicAdd(&cnt[i0], 1);
        tok_idx[i0 * TT + p] = wave; tok_prob[i0 * TT + p] = p0;
        p = atomicAdd(&cnt[i1], 1);
        tok_idx[i1 * TT + p] = wave; tok_prob[i1 * TT + p] = p1;
    }
}

// ---------------------------------------------------------------------------
// Fused expert FFN: per block, 32 gathered tokens x full D output.
// 512 threads = 16 waves; wave w owns D columns [w*64, w*64+64).
// X rows staged via global_load_async_to_lds_b128 (ASYNCcnt path), then
// converted fp32->bf16 in LDS. F loop (FC=128): h = gelu(x@W1chunk) -> LDS;
// acc += h@W2chunk with software-pipelined, sched_group_barrier-ordered loads.
// W1t: [E][F][D] bf16 (B-frag contiguous in K=d); W2t: [E][D][F] bf16.
// ---------------------------------------------------------------------------
__global__ __launch_bounds__(512, 1)
void moe_ffn_kernel(const float* __restrict__ x,
                    const bf16* __restrict__ W1t,
                    const float* __restrict__ b1,
                    const bf16* __restrict__ W2t,
                    const float* __restrict__ b2,
                    const int* __restrict__ cnt,
                    const int* __restrict__ tok_idx,
                    const float* __restrict__ tok_prob,
                    float* __restrict__ out) {
    const int e   = blockIdx.x / MAXBLK;
    const int blk = blockIdx.x % MAXBLK;
    const int nt  = cnt[e];
    if (blk * TM >= nt) return;

    __shared__ bf16 xs[TM][XS_LD];
    // 64 KB pool: fp32 async-staging buffer in the prologue, hs afterwards.
    __shared__ __align__(16) char smem_pool[16 * DD * sizeof(float)];
    float (*xstage)[DD]  = reinterpret_cast<float (*)[DD]>(smem_pool);
    bf16  (*hs)[HS_LD]   = reinterpret_cast<bf16 (*)[HS_LD]>(smem_pool);
    __shared__ int   tok_s[TM];
    __shared__ float prb_s[TM];

    const int tid  = threadIdx.x;
    const int lane = tid & 31;
    const int wv   = tid >> 5;      // 0..15
    const int mlo  = lane & 15;     // N column within 16-wide tile / A row
    const int khalf = lane >> 4;    // which K-half this lane holds

    if (tid < TM) {
        const int slot = blk * TM + tid;
        tok_s[tid] = (slot < nt) ? tok_idx[e * TT + slot] : 0;
        prb_s[tid] = (slot < nt) ? tok_prob[e * TT + slot] : 0.f;
    }
    __syncthreads();

    // ---- Stage gathered X tile: async global->LDS (fp32), then cvt to bf16.
    // Each wave owns one staging row per half; per-wave s_wait_asynccnt is
    // sufficient (no cross-wave hazard on xstage[wv]).
    #pragma unroll
    for (int half = 0; half < 2; ++half) {
        const int r = half * 16 + wv;
        const float* xr = x + (size_t)tok_s[r] * DD;
        #pragma unroll
        for (int it = 0; it < 8; ++it) {
            const uint32_t ldsa =
                (uint32_t)(uintptr_t)&xstage[wv][it * 128 + lane * 4];
            const float* ga = xr + it * 128 + lane * 4;
            asm volatile("global_load_async_to_lds_b128 %0, %1, off"
                         :: "v"(ldsa), "v"(ga) : "memory");
        }
        asm volatile("s_wait_asynccnt 0x0" ::: "memory");
        #pragma unroll
        for (int d4 = lane * 4; d4 < DD; d4 += 32 * 4) {
            const float4 v = *(const float4*)&xstage[wv][d4];
            *(v4bf*)(&xs[r][d4]) =
                (v4bf){(bf16)v.x, (bf16)v.y, (bf16)v.z, (bf16)v.w};
        }
    }
    __syncthreads();   // xs complete; xstage dead -> pool reused as hs

    const bf16*  W1e = W1t + (size_t)e * FF * DD;
    const bf16*  W2e = W2t + (size_t)e * DD * FF;
    const float* b1e = b1 + (size_t)e * FF;
    const float* b2e = b2 + (size_t)e * DD;

    v8f acc[2][4];
    #pragma unroll
    for (int rt = 0; rt < 2; ++rt)
        #pragma unroll
        for (int ct = 0; ct < 4; ++ct) acc[rt][ct] = (v8f){};

    for (int f0 = 0; f0 < FF; f0 += FC) {
        // ---- Stage A: one 16x16 h-tile per wave (2x8 tile grid over 32xFC)
        {
            const int rt = wv >> 3, ct = wv & 7;
            const int fcol = f0 + ct * 16 + mlo;            // lane's N (f index)
            const bf16* wrow = W1e + (size_t)fcol * DD;     // contiguous in K=d
            const bf16* ar   = &xs[rt * 16 + mlo][0];
            if (f0 + FC < FF)   // warm L2 for next chunk (global_prefetch_b8)
                __builtin_prefetch((const void*)(wrow + (size_t)FC * DD), 0, 1);
            v8f c = (v8f){};
            v16bf bnext = *(const v16bf*)(wrow + khalf * 16);  // k0 = 0
            #pragma unroll
            for (int k0 = 0; k0 < DD; k0 += 32) {
                const v16bf bfrag = bnext;
                if (k0 + 32 < DD)
                    bnext = *(const v16bf*)(wrow + k0 + 32 + khalf * 16);
                const v16bf a = cat8(*(const v8bf*)(ar + k0 + khalf * 16),
                                     *(const v8bf*)(ar + k0 + khalf * 16 + 8));
                c = __builtin_amdgcn_wmma_f32_16x16x32_bf16(
                        false, a, false, bfrag, (short)0, c, false, false);
                // Keep next-step W1 loads ahead of this step's WMMA.
                __builtin_amdgcn_sched_group_barrier(SG_VMREAD, 2, 0);
                __builtin_amdgcn_sched_group_barrier(SG_DSREAD, 2, 0);
                __builtin_amdgcn_sched_group_barrier(SG_WMMA,   1, 0);
            }
            const float bias = b1e[fcol];
            #pragma unroll
            for (int v = 0; v < 8; ++v) {
                const float hv = c[v] + bias;
                const float g  = 0.5f * hv * (1.f + erff(hv * 0.70710678118f));
                hs[rt * 16 + 8 * khalf + v][ct * 16 + mlo] = (bf16)g;
            }
        }
        __syncthreads();

        // ---- Stage B: acc(32x64 per wave) += h(32xFC) @ W2chunk(FCx64)
        // Software pipeline (distance 1): next K-step's 8 global b128 + 4 ds
        // b128 loads issue before the current step's 8 WMMAs.
        {
            const int d0 = wv * 64;
            const bf16* w2base = W2e + (size_t)(d0 + mlo) * FF + f0 + khalf * 16;
            const bf16* ar0 = &hs[mlo][0];
            const bf16* ar1 = &hs[16 + mlo][0];
            if (f0 + FC < FF)
                __builtin_prefetch((const void*)(w2base + FC), 0, 1);

            v16bf b0  = *(const v16bf*)(w2base + (size_t)(0 * 16) * FF);
            v16bf b1f = *(const v16bf*)(w2base + (size_t)(1 * 16) * FF);
            v16bf b2f = *(const v16bf*)(w2base + (size_t)(2 * 16) * FF);
            v16bf b3f = *(const v16bf*)(w2base + (size_t)(3 * 16) * FF);
            v16bf a0 = cat8(*(const v8bf*)(ar0 + khalf * 16),
                            *(const v8bf*)(ar0 + khalf * 16 + 8));
            v16bf a1 = cat8(*(const v8bf*)(ar1 + khalf * 16),
                            *(const v8bf*)(ar1 + khalf * 16 + 8));

            #pragma unroll
            for (int ks = 0; ks < FC; ks += 32) {
                v16bf nb0 = b0, nb1 = b1f, nb2 = b2f, nb3 = b3f;
                v16bf na0 = a0, na1 = a1;
                if (ks + 32 < FC) {
                    const int kn = ks + 32;
                    nb0 = *(const v16bf*)(w2base + (size_t)(0 * 16) * FF + kn);
                    nb1 = *(const v16bf*)(w2base + (size_t)(1 * 16) * FF + kn);
                    nb2 = *(const v16bf*)(w2base + (size_t)(2 * 16) * FF + kn);
                    nb3 = *(const v16bf*)(w2base + (size_t)(3 * 16) * FF + kn);
                    na0 = cat8(*(const v8bf*)(ar0 + kn + khalf * 16),
                               *(const v8bf*)(ar0 + kn + khalf * 16 + 8));
                    na1 = cat8(*(const v8bf*)(ar1 + kn + khalf * 16),
                               *(const v8bf*)(ar1 + kn + khalf * 16 + 8));
                }
                acc[0][0] = __builtin_amdgcn_wmma_f32_16x16x32_bf16(
                    false, a0, false, b0, (short)0, acc[0][0], false, false);
                acc[1][0] = __builtin_amdgcn_wmma_f32_16x16x32_bf16(
                    false, a1, false, b0, (short)0, acc[1][0], false, false);
                acc[0][1] = __builtin_amdgcn_wmma_f32_16x16x32_bf16(
                    false, a0, false, b1f, (short)0, acc[0][1], false, false);
                acc[1][1] = __builtin_amdgcn_wmma_f32_16x16x32_bf16(
                    false, a1, false, b1f, (short)0, acc[1][1], false, false);
                acc[0][2] = __builtin_amdgcn_wmma_f32_16x16x32_bf16(
                    false, a0, false, b2f, (short)0, acc[0][2], false, false);
                acc[1][2] = __builtin_amdgcn_wmma_f32_16x16x32_bf16(
                    false, a1, false, b2f, (short)0, acc[1][2], false, false);
                acc[0][3] = __builtin_amdgcn_wmma_f32_16x16x32_bf16(
                    false, a0, false, b3f, (short)0, acc[0][3], false, false);
                acc[1][3] = __builtin_amdgcn_wmma_f32_16x16x32_bf16(
                    false, a1, false, b3f, (short)0, acc[1][3], false, false);
                // Per pipeline step: next loads first, then this step's WMMAs.
                __builtin_amdgcn_sched_group_barrier(SG_VMREAD, 8, 0);
                __builtin_amdgcn_sched_group_barrier(SG_DSREAD, 4, 0);
                __builtin_amdgcn_sched_group_barrier(SG_WMMA,   8, 0);
                b0 = nb0; b1f = nb1; b2f = nb2; b3f = nb3;
                a0 = na0; a1 = na1;
            }
        }
        __syncthreads();
    }

    // ---- Weighted combine into out (atomic add; inactive rows have p=0)
    const int d0 = wv * 64;
    #pragma unroll
    for (int ct = 0; ct < 4; ++ct) {
        const int dcol = d0 + ct * 16 + mlo;
        const float b2v = b2e[dcol];
        #pragma unroll
        for (int rt = 0; rt < 2; ++rt) {
            #pragma unroll
            for (int v = 0; v < 8; ++v) {
                const int r = rt * 16 + 8 * khalf + v;
                const float val = prb_s[r] * (acc[rt][ct][v] + b2v);
                atomicAdd(&out[(size_t)tok_s[r] * DD + dcol], val);
            }
        }
    }
}

// ---------------------------------------------------------------------------
extern "C" void kernel_launch(void* const* d_in, const int* in_sizes, int n_in,
                              void* d_out, int out_size, void* d_ws, size_t ws_size,
                              hipStream_t stream) {
    const float* x      = (const float*)d_in[0]; // [B,S,D]
    const float* gate_w = (const float*)d_in[1]; // [D,E]
    const float* w1     = (const float*)d_in[2]; // [E,D,F]
    const float* b1     = (const float*)d_in[3]; // [E,F]
    const float* w2     = (const float*)d_in[4]; // [E,F,D]
    const float* b2     = (const float*)d_in[5]; // [E,D]
    float* out = (float*)d_out;                  // [B,S,D]

    // Workspace layout
    char* ws = (char*)d_ws;
    bf16* W1t = (bf16*)ws; ws += (size_t)EE * DD * FF * sizeof(bf16); // [E][F][D]
    bf16* W2t = (bf16*)ws; ws += (size_t)EE * DD * FF * sizeof(bf16); // [E][D][F]
    int*   cnt      = (int*)ws;   ws += 256;
    int*   tok_idx  = (int*)ws;   ws += (size_t)EE * TT * sizeof(int);
    float* tok_prob = (float*)ws; ws += (size_t)EE * TT * sizeof(float);

    hipMemsetAsync(out, 0, (size_t)out_size * sizeof(float), stream);
    hipMemsetAsync(cnt, 0, EE * sizeof(int), stream);

    // Weight transpose+convert: bf16 weights (134 MB) fit the 192 MB L2.
    {
        dim3 g1(FF / 32, DD / 32, EE);  // w1 [E][D][F] -> W1t [E][F][D]
        transpose_bf16_kernel<<<g1, 256, 0, stream>>>(w1, W1t, DD, FF);
        dim3 g2(DD / 32, FF / 32, EE);  // w2 [E][F][D] -> W2t [E][D][F]
        transpose_bf16_kernel<<<g2, 256, 0, stream>>>(w2, W2t, FF, DD);
    }

    // Router: one wave per token; 8 waves/block.
    router_kernel<<<TT / 8, 256, 0, stream>>>(x, gate_w, cnt, tok_idx, tok_prob);

    // Expert FFN: worst-case grid; idle blocks exit early on cnt[e].
    moe_ffn_kernel<<<EE * MAXBLK, 512, 0, stream>>>(
        x, W1t, b1, W2t, b2, cnt, tok_idx, tok_prob, out);
}
